// E_HypergraphConv_51196010168982
// MI455X (gfx1250) — matrix operand order repaired and checked by reading, not compiled, not run on hardware
//
#include <hip/hip_runtime.h>
#include <hip/hip_bf16.h>

// ---------------------------------------------------------------------------
// E_HypergraphConv for MI455X (gfx1250, wave32, WMMA f16 -> f32 accumulate)
// ---------------------------------------------------------------------------

typedef __attribute__((ext_vector_type(16))) _Float16 v16h;
typedef __attribute__((ext_vector_type(8)))  float    v8f;

#define N_NODES  50000
#define H_EDGES  10000
#define NNZ_CNT  400000
#define T_TOT    (N_NODES + H_EDGES)      // 60000
#define M_EDGES  (2 * NNZ_CNT)            // 800000
#define DIM      64

__device__ __forceinline__ void lds_fence() {
    // DS ops are in-order per wave; dscnt==0 makes all prior LDS stores
    // visible to subsequent LDS loads of this wave. "memory" stops reordering.
    asm volatile("s_wait_dscnt 0" ::: "memory");
}

__device__ __forceinline__ v8f vzero8() {
    v8f v = {0.f, 0.f, 0.f, 0.f, 0.f, 0.f, 0.f, 0.f};
    return v;
}

__device__ __forceinline__ float silu(float x) {
    return x / (1.0f + expf(-x));
}

// A-matrix (16x32 f16) element->K mapping (CDNA5 ISA 7.12.2)
__device__ __forceinline__ int a_k(int e, int lane16) {
    return ((e < 8) ? e : (e + 8)) + (lane16 ? 8 : 0);
}

// ---------------------------------------------------------------------------
// Kernel 1: scatter node coords into hyperedge accumulators + degree counts
// ---------------------------------------------------------------------------
__global__ void scatter_coord_kernel(const int* __restrict__ node_idx,
                                     const int* __restrict__ hyper_idx,
                                     const float* __restrict__ node_coord,
                                     float* __restrict__ coordW,
                                     int* __restrict__ cnt) {
    int i = blockIdx.x * blockDim.x + threadIdx.x;
    if (i >= NNZ_CNT) return;
    int nidx = node_idx[i];
    int hidx = hyper_idx[i];
    float cx = node_coord[nidx * 3 + 0];
    float cy = node_coord[nidx * 3 + 1];
    float cz = node_coord[nidx * 3 + 2];
    atomicAdd(&coordW[(size_t)(N_NODES + hidx) * 3 + 0], cx);
    atomicAdd(&coordW[(size_t)(N_NODES + hidx) * 3 + 1], cy);
    atomicAdd(&coordW[(size_t)(N_NODES + hidx) * 3 + 2], cz);
    atomicAdd(&cnt[nidx], 1);
    atomicAdd(&cnt[N_NODES + hidx], 1);
}

// ---------------------------------------------------------------------------
// Kernel 2: coordW[0:N] = node_coord; coordW[N:T] /= max(cnt,1)
// ---------------------------------------------------------------------------
__global__ void finalize_coord_kernel(const float* __restrict__ node_coord,
                                      const int* __restrict__ cnt,
                                      float* __restrict__ coordW) {
    int t = blockIdx.x * blockDim.x + threadIdx.x;
    if (t >= T_TOT) return;
    if (t < N_NODES) {
        coordW[t * 3 + 0] = node_coord[t * 3 + 0];
        coordW[t * 3 + 1] = node_coord[t * 3 + 1];
        coordW[t * 3 + 2] = node_coord[t * 3 + 2];
    } else {
        float c = (float)max(cnt[t], 1);
        coordW[t * 3 + 0] /= c;
        coordW[t * 3 + 1] /= c;
        coordW[t * 3 + 2] /= c;
    }
}

// ---------------------------------------------------------------------------
// Kernel 3: h[0:N] = x @ lin_w.T + lin_b  (WMMA). Writes f32 h and f16 h16.
// One wave per 16-row tile, 8 waves/block.
// ---------------------------------------------------------------------------
__global__ void lin_kernel(const float* __restrict__ x,
                           const float* __restrict__ lin_w,
                           const float* __restrict__ lin_b,
                           float* __restrict__ h,
                           _Float16* __restrict__ h16) {
    __shared__ _Float16 sW[64 * 64];
    __shared__ float    sB[64];
    int tid = threadIdx.x;
    for (int i = tid; i < 64 * 64; i += 256) sW[i] = (_Float16)lin_w[i];
    if (tid < 64) sB[tid] = lin_b[tid];
    __syncthreads();

    int wave = tid >> 5, lane = tid & 31, lane16 = lane >> 4, n = lane & 15;
    int t0 = (blockIdx.x * 8 + wave) * 16;
    if (t0 >= N_NODES) return;

    int row = t0 + n;
    int rowc = (row < N_NODES) ? row : (N_NODES - 1);
    const float* xr = x + (size_t)rowc * DIM;

    v8f acc[4];
#pragma unroll
    for (int nt = 0; nt < 4; nt++) acc[nt] = vzero8();

#pragma unroll
    for (int c = 0; c < 2; c++) {
        v16h a;
#pragma unroll
        for (int e = 0; e < 16; e++) {
            int f = 32 * c + a_k(e, lane16);
            a[e] = (_Float16)xr[f];
        }
#pragma unroll
        for (int nt = 0; nt < 4; nt++) {
            v16h b;
#pragma unroll
            for (int e = 0; e < 16; e++) {
                int k = 32 * c + a_k(e, lane16);
                b[e] = sW[(nt * 16 + n) * 64 + k];
            }
            acc[nt] = __builtin_amdgcn_wmma_f32_16x16x32_f16(
                false, a, false, b, (short)0, acc[nt], false, false);
        }
    }
#pragma unroll
    for (int nt = 0; nt < 4; nt++) {
        int col = nt * 16 + n;
        float bb = sB[col];
#pragma unroll
        for (int r = 0; r < 8; r++) {
            int grow = t0 + r + lane16 * 8;
            if (grow < N_NODES) {
                float v = acc[nt][r] + bb;
                h[(size_t)grow * DIM + col]   = v;
                h16[(size_t)grow * DIM + col] = (_Float16)v;
            }
        }
    }
}

// ---------------------------------------------------------------------------
// Kernel 4: h[N:T] = hyperedge_feature (f32 + f16 copies)
// ---------------------------------------------------------------------------
__global__ void copy_hyper_kernel(const float* __restrict__ hf,
                                  float* __restrict__ h,
                                  _Float16* __restrict__ h16) {
    int i = blockIdx.x * blockDim.x + threadIdx.x;
    if (i >= H_EDGES * DIM) return;
    float v = hf[i];
    h[(size_t)N_NODES * DIM + i]   = v;
    h16[(size_t)N_NODES * DIM + i] = (_Float16)v;
}

// ---------------------------------------------------------------------------
// Kernel 5: fused edge pipeline. One wave per 16-edge tile.
//   gather h16[row],h16[col], RBF(radial) -> [16x192]
//   ef  = silu(silu(feat@ew1.T+eb1)@ew2.T+eb2)
//   agg[row]   += ef                       (atomic f32)
//   cm  = silu(ef@cw1.T+cb1); dot = cm@cw2
//   trans[row] += cd * tanh(dot)           (atomic f32)
// ---------------------------------------------------------------------------
__global__ void edge_kernel(const int* __restrict__ node_idx,
                            const int* __restrict__ hyper_idx,
                            const float* __restrict__ coordW,
                            const _Float16* __restrict__ h16,
                            const float* __restrict__ ew1, const float* __restrict__ eb1,
                            const float* __restrict__ ew2, const float* __restrict__ eb2,
                            const float* __restrict__ cw1, const float* __restrict__ cb1,
                            const float* __restrict__ cw2,
                            const float* __restrict__ rbf_centers,
                            const float* __restrict__ rbf_gamma,
                            const float* __restrict__ rbf_w,
                            const float* __restrict__ rbf_b,
                            float* __restrict__ agg,
                            float* __restrict__ trans) {
    __shared__ _Float16 sW1[192 * 64];   // ew1 [out=64][in=192]
    __shared__ _Float16 sW2[64 * 64];    // ew2
    __shared__ _Float16 sCW[64 * 64];    // cw1
    __shared__ float sB1[64], sB2[64], sCB[64], sCW2v[64], sRB[64];
    __shared__ float sRW[128];
    __shared__ _Float16 stage[8][16 * 64];
    __shared__ int   rowIdxS[8][16];
    __shared__ float dotS[8][16];

    int tid = threadIdx.x;
    for (int i = tid; i < 192 * 64; i += 256) sW1[i] = (_Float16)ew1[i];
    for (int i = tid; i < 64 * 64; i += 256) { sW2[i] = (_Float16)ew2[i]; sCW[i] = (_Float16)cw1[i]; }
    if (tid < 64) {
        sB1[tid] = eb1[tid]; sB2[tid] = eb2[tid]; sCB[tid] = cb1[tid];
        sCW2v[tid] = cw2[tid]; sRB[tid] = rbf_b[tid];
    }
    if (tid < 128) sRW[tid] = rbf_w[tid];
    __syncthreads();

    int wave = tid >> 5, lane = tid & 31, lane16 = lane >> 4, n = lane & 15;
    long m0 = ((long)blockIdx.x * 8 + wave) * 16;

    // each lane (and its +16 partner) owns edge em = lane&15 of the tile
    long m = m0 + n;
    bool mval = (m < (long)M_EDGES);
    long mc = mval ? m : (long)(M_EDGES - 1);
    int r_idx, c_idx;
    if (mc < (long)NNZ_CNT) {
        r_idx = node_idx[mc];
        c_idx = N_NODES + hyper_idx[mc];
    } else {
        int j = (int)(mc - NNZ_CNT);
        r_idx = N_NODES + hyper_idx[j];
        c_idx = node_idx[j];
    }
    if (lane < 16) rowIdxS[wave][lane] = mval ? r_idx : -1;

    // coord2radial + unit direction
    float dx = coordW[(size_t)r_idx * 3 + 0] - coordW[(size_t)c_idx * 3 + 0];
    float dy = coordW[(size_t)r_idx * 3 + 1] - coordW[(size_t)c_idx * 3 + 1];
    float dz = coordW[(size_t)r_idx * 3 + 2] - coordW[(size_t)c_idx * 3 + 2];
    float radial = dx * dx + dy * dy + dz * dz;
    float nr = sqrtf(radial) + 1e-8f;
    float cdx = dx / nr, cdy = dy / nr, cdz = dz / nr;

    // RBF scalars (2 centers)
    float g  = rbf_gamma[0];
    float d0 = fabsf(radial - rbf_centers[0]);
    float d1 = fabsf(radial - rbf_centers[1]);
    float e0 = expf(-g * d0 * d0);
    float e1 = expf(-g * d1 * d1);

    const _Float16* hr = h16 + (size_t)r_idx * DIM;
    const _Float16* hc = h16 + (size_t)c_idx * DIM;

    // ---------------- edge MLP layer 1: [16x192] @ [192x64] ----------------
    v8f acc[4];
#pragma unroll
    for (int nt = 0; nt < 4; nt++) acc[nt] = vzero8();

    for (int c = 0; c < 6; c++) {
        v16h a;
#pragma unroll
        for (int e = 0; e < 16; e++) {
            int f = 32 * c + a_k(e, lane16);
            float v;
            if (f < 64)       v = (float)hr[f];
            else if (f < 128) v = (float)hc[f - 64];
            else {
                int j = f - 128;
                v = e0 * sRW[2 * j] + e1 * sRW[2 * j + 1] + sRB[j];
            }
            a[e] = (_Float16)v;
        }
#pragma unroll
        for (int nt = 0; nt < 4; nt++) {
            v16h b;
#pragma unroll
            for (int e = 0; e < 16; e++)
                b[e] = sW1[(nt * 16 + n) * 192 + 32 * c + a_k(e, lane16)];
            acc[nt] = __builtin_amdgcn_wmma_f32_16x16x32_f16(
                false, a, false, b, (short)0, acc[nt], false, false);
        }
    }
    _Float16* st = stage[wave];
#pragma unroll
    for (int nt = 0; nt < 4; nt++) {
        float bb = sB1[nt * 16 + n];
#pragma unroll
        for (int r = 0; r < 8; r++)
            st[(r + lane16 * 8) * 64 + nt * 16 + n] = (_Float16)silu(acc[nt][r] + bb);
    }
    lds_fence();

    // ---------------- edge MLP layer 2: [16x64] @ [64x64] ------------------
    v8f acc2[4];
#pragma unroll
    for (int nt = 0; nt < 4; nt++) acc2[nt] = vzero8();
#pragma unroll
    for (int c = 0; c < 2; c++) {
        v16h a;
#pragma unroll
        for (int e = 0; e < 16; e++)
            a[e] = st[n * 64 + 32 * c + a_k(e, lane16)];
#pragma unroll
        for (int nt = 0; nt < 4; nt++) {
            v16h b;
#pragma unroll
            for (int e = 0; e < 16; e++)
                b[e] = sW2[(nt * 16 + n) * 64 + 32 * c + a_k(e, lane16)];
            acc2[nt] = __builtin_amdgcn_wmma_f32_16x16x32_f16(
                false, a, false, b, (short)0, acc2[nt], false, false);
        }
    }
    lds_fence();  // done reading stage before overwriting it with ef

    // ef values: scatter into agg (atomics) + restage for coord MLP
#pragma unroll
    for (int nt = 0; nt < 4; nt++) {
        int col = nt * 16 + n;
        float bb = sB2[col];
#pragma unroll
        for (int r = 0; r < 8; r++) {
            int mrow = r + lane16 * 8;
            float v = silu(acc2[nt][r] + bb);
            st[mrow * 64 + col] = (_Float16)v;
            int rg = rowIdxS[wave][mrow];
            if (rg >= 0) atomicAdd(&agg[(size_t)rg * DIM + col], v);
        }
    }
    lds_fence();

    // ---------------- coord MLP: cm = silu(ef@cw1.T+cb1) -------------------
    v8f acc3[4];
#pragma unroll
    for (int nt = 0; nt < 4; nt++) acc3[nt] = vzero8();
#pragma unroll
    for (int c = 0; c < 2; c++) {
        v16h a;
#pragma unroll
        for (int e = 0; e < 16; e++)
            a[e] = st[n * 64 + 32 * c + a_k(e, lane16)];
#pragma unroll
        for (int nt = 0; nt < 4; nt++) {
            v16h b;
#pragma unroll
            for (int e = 0; e < 16; e++)
                b[e] = sCW[(nt * 16 + n) * 64 + 32 * c + a_k(e, lane16)];
            acc3[nt] = __builtin_amdgcn_wmma_f32_16x16x32_f16(
                false, a, false, b, (short)0, acc3[nt], false, false);
        }
    }

    // dot = cm @ cw2 : per-lane partials then xor-shuffle reduce over 16 lanes
    float p[8];
#pragma unroll
    for (int r = 0; r < 8; r++) {
        float s = 0.f;
#pragma unroll
        for (int nt = 0; nt < 4; nt++) {
            int col = nt * 16 + n;
            s += silu(acc3[nt][r] + sCB[col]) * sCW2v[col];
        }
        p[r] = s;
    }
#pragma unroll
    for (int r = 0; r < 8; r++) {
        for (int msk = 1; msk < 16; msk <<= 1)
            p[r] += __shfl_xor(p[r], msk, 32);
    }
    if (n == 0) {
#pragma unroll
        for (int r = 0; r < 8; r++) dotS[wave][r + lane16 * 8] = p[r];
    }
    lds_fence();

    // trans scatter (lanes 0..15 each own one edge of the tile)
    if (lane < 16 && mval) {
        float t = tanhf(dotS[wave][lane]);
        atomicAdd(&trans[(size_t)r_idx * 3 + 0], cdx * t);
        atomicAdd(&trans[(size_t)r_idx * 3 + 1], cdy * t);
        atomicAdd(&trans[(size_t)r_idx * 3 + 2], cdz * t);
    }
}

// ---------------------------------------------------------------------------
// Kernel 6: coord output = coordW + trans / max(cnt,1)
// ---------------------------------------------------------------------------
__global__ void coord_out_kernel(const float* __restrict__ coordW,
                                 const float* __restrict__ trans,
                                 const int* __restrict__ cnt,
                                 float* __restrict__ outC) {
    int t = blockIdx.x * blockDim.x + threadIdx.x;
    if (t >= T_TOT) return;
    float c = (float)max(cnt[t], 1);
    outC[t * 3 + 0] = coordW[t * 3 + 0] + trans[t * 3 + 0] / c;
    outC[t * 3 + 1] = coordW[t * 3 + 1] + trans[t * 3 + 1] / c;
    outC[t * 3 + 2] = coordW[t * 3 + 2] + trans[t * 3 + 2] / c;
}

// ---------------------------------------------------------------------------
// Kernel 7: node model. out = h + silu(concat(h,agg)@nw1.T+nb1)@nw2.T+nb2
// ---------------------------------------------------------------------------
__global__ void node_kernel(const float* __restrict__ h,
                            const float* __restrict__ agg,
                            const float* __restrict__ nw1, const float* __restrict__ nb1,
                            const float* __restrict__ nw2, const float* __restrict__ nb2,
                            float* __restrict__ outH) {
    __shared__ _Float16 sW1[128 * 64];   // nw1 [out=64][in=128]
    __shared__ _Float16 sW2[64 * 64];
    __shared__ float sB1[64], sB2[64];
    __shared__ _Float16 stage[8][16 * 64];

    int tid = threadIdx.x;
    for (int i = tid; i < 128 * 64; i += 256) sW1[i] = (_Float16)nw1[i];
    for (int i = tid; i < 64 * 64; i += 256) sW2[i] = (_Float16)nw2[i];
    if (tid < 64) { sB1[tid] = nb1[tid]; sB2[tid] = nb2[tid]; }
    __syncthreads();

    int wave = tid >> 5, lane = tid & 31, lane16 = lane >> 4, n = lane & 15;
    int t0 = (blockIdx.x * 8 + wave) * 16;
    if (t0 >= T_TOT) return;

    int tr = t0 + n;
    int trc = (tr < T_TOT) ? tr : (T_TOT - 1);
    const float* hrow = h + (size_t)trc * DIM;
    const float* arow = agg + (size_t)trc * DIM;

    v8f acc[4];
#pragma unroll
    for (int nt = 0; nt < 4; nt++) acc[nt] = vzero8();
#pragma unroll
    for (int c = 0; c < 4; c++) {
        v16h a;
#pragma unroll
        for (int e = 0; e < 16; e++) {
            int f = 32 * c + a_k(e, lane16);
            a[e] = (_Float16)((f < 64) ? hrow[f] : arow[f - 64]);
        }
#pragma unroll
        for (int nt = 0; nt < 4; nt++) {
            v16h b;
#pragma unroll
            for (int e = 0; e < 16; e++)
                b[e] = sW1[(nt * 16 + n) * 128 + 32 * c + a_k(e, lane16)];
            acc[nt] = __builtin_amdgcn_wmma_f32_16x16x32_f16(
                false, a, false, b, (short)0, acc[nt], false, false);
        }
    }
    _Float16* st = stage[wave];
#pragma unroll
    for (int nt = 0; nt < 4; nt++) {
        float bb = sB1[nt * 16 + n];
#pragma unroll
        for (int r = 0; r < 8; r++)
            st[(r + lane16 * 8) * 64 + nt * 16 + n] = (_Float16)silu(acc[nt][r] + bb);
    }
    lds_fence();

    v8f acc2[4];
#pragma unroll
    for (int nt = 0; nt < 4; nt++) acc2[nt] = vzero8();
#pragma unroll
    for (int c = 0; c < 2; c++) {
        v16h a;
#pragma unroll
        for (int e = 0; e < 16; e++)
            a[e] = st[n * 64 + 32 * c + a_k(e, lane16)];
#pragma unroll
        for (int nt = 0; nt < 4; nt++) {
            v16h b;
#pragma unroll
            for (int e = 0; e < 16; e++)
                b[e] = sW2[(nt * 16 + n) * 64 + 32 * c + a_k(e, lane16)];
            acc2[nt] = __builtin_amdgcn_wmma_f32_16x16x32_f16(
                false, a, false, b, (short)0, acc2[nt], false, false);
        }
    }
#pragma unroll
    for (int nt = 0; nt < 4; nt++) {
        int col = nt * 16 + n;
        float bb = sB2[col];
#pragma unroll
        for (int r = 0; r < 8; r++) {
            int grow = t0 + r + lane16 * 8;
            if (grow < T_TOT) {
                size_t off = (size_t)grow * DIM + col;
                outH[off] = h[off] + acc2[nt][r] + bb;
            }
        }
    }
}

// ---------------------------------------------------------------------------
// Host launcher
// ---------------------------------------------------------------------------
extern "C" void kernel_launch(void* const* d_in, const int* in_sizes, int n_in,
                              void* d_out, int out_size, void* d_ws, size_t ws_size,
                              hipStream_t stream) {
    const float* x       = (const float*)d_in[0];
    const float* hf      = (const float*)d_in[1];
    const int*   nidx    = (const int*)d_in[2];
    const int*   hidx    = (const int*)d_in[3];
    const float* ncoord  = (const float*)d_in[4];
    const float* lin_w   = (const float*)d_in[5];
    const float* lin_b   = (const float*)d_in[6];
    const float* ew1     = (const float*)d_in[7];
    const float* eb1     = (const float*)d_in[8];
    const float* ew2     = (const float*)d_in[9];
    const float* eb2     = (const float*)d_in[10];
    const float* cw1     = (const float*)d_in[11];
    const float* cb1     = (const float*)d_in[12];
    const float* cw2     = (const float*)d_in[13];
    const float* nw1     = (const float*)d_in[14];
    const float* nb1     = (const float*)d_in[15];
    const float* nw2     = (const float*)d_in[16];
    const float* nb2     = (const float*)d_in[17];
    const float* rbf_c   = (const float*)d_in[18];
    const float* rbf_g   = (const float*)d_in[19];
    const float* rbf_w   = (const float*)d_in[20];
    const float* rbf_b   = (const float*)d_in[21];

    // workspace layout
    float* h      = (float*)d_ws;                          // T*64
    float* agg    = h      + (size_t)T_TOT * DIM;          // T*64
    float* coordW = agg    + (size_t)T_TOT * DIM;          // T*3
    float* trans  = coordW + (size_t)T_TOT * 3;            // T*3
    int*   cnt    = (int*)(trans + (size_t)T_TOT * 3);     // T
    _Float16* h16 = (_Float16*)(cnt + T_TOT);              // T*64 halves

    float* outH = (float*)d_out;                           // T*64 (h[:N] then h[N:])
    float* outC = outH + (size_t)T_TOT * DIM;              // T*3

    hipMemsetAsync(agg,    0, (size_t)T_TOT * DIM * sizeof(float), stream);
    hipMemsetAsync(coordW, 0, (size_t)T_TOT * 3 * sizeof(float), stream);
    hipMemsetAsync(trans,  0, (size_t)T_TOT * 3 * sizeof(float), stream);
    hipMemsetAsync(cnt,    0, (size_t)T_TOT * sizeof(int), stream);

    scatter_coord_kernel<<<(NNZ_CNT + 255) / 256, 256, 0, stream>>>(
        nidx, hidx, ncoord, coordW, cnt);

    finalize_coord_kernel<<<(T_TOT + 255) / 256, 256, 0, stream>>>(
        ncoord, cnt, coordW);

    {
        int tiles = (N_NODES + 15) / 16;
        lin_kernel<<<(tiles + 7) / 8, 256, 0, stream>>>(x, lin_w, lin_b, h, h16);
    }

    copy_hyper_kernel<<<(H_EDGES * DIM + 255) / 256, 256, 0, stream>>>(hf, h, h16);

    {
        int tiles = (M_EDGES + 15) / 16;
        edge_kernel<<<(tiles + 7) / 8, 256, 0, stream>>>(
            nidx, hidx, coordW, h16,
            ew1, eb1, ew2, eb2, cw1, cb1, cw2,
            rbf_c, rbf_g, rbf_w, rbf_b,
            agg, trans);
    }

    coord_out_kernel<<<(T_TOT + 255) / 256, 256, 0, stream>>>(coordW, trans, cnt, outC);

    {
        int tiles = (T_TOT + 15) / 16;
        node_kernel<<<(tiles + 7) / 8, 256, 0, stream>>>(h, agg, nw1, nb1, nw2, nb2, outH);
    }
}
